// GAT_14173392077063
// MI455X (gfx1250) — compile-verified, matrix operand order checked by reading, not run on hardware
//
#include <hip/hip_runtime.h>

#define NEG_SLOPE 0.2f
#define BN_EPS    1e-5f

typedef _Float16 v16h  __attribute__((ext_vector_type(16)));
typedef _Float16 half8 __attribute__((ext_vector_type(8)));
typedef float    v8f   __attribute__((ext_vector_type(8)));

// ---------------------------------------------------------------------------
// f32 -> f16 conversion (grid-stride not needed; exact launch)
// ---------------------------------------------------------------------------
__global__ void k_cvt_f16(const float* __restrict__ x, _Float16* __restrict__ y, long long n) {
    long long i = (long long)blockIdx.x * blockDim.x + threadIdx.x;
    if (i < n) y[i] = (_Float16)x[i];
}

// Convert weight W [K x Ncols] (row-major f32) to transposed f16 WT [Ncols x K]
__global__ void k_cvt_wT(const float* __restrict__ W, _Float16* __restrict__ WT, int K, int Ncols) {
    int i = blockIdx.x * blockDim.x + threadIdx.x;
    if (i >= K * Ncols) return;
    int k = i / Ncols;
    int n = i - k * Ncols;
    WT[(size_t)n * K + k] = (_Float16)W[i];
}

// ---------------------------------------------------------------------------
// WMMA GEMM: C[M x Ncols] = A[M x K] * B[K x Ncols], B given transposed (BT: Ncols x K)
// A, BT are f16; C is f32. One wave computes one 16x16 tile; 4 waves / block.
// M is a multiple of 16 (N=100000=6250*16). K multiple of 32.
// ---------------------------------------------------------------------------
__global__ void k_gemm_wmma(const _Float16* __restrict__ A, const _Float16* __restrict__ BT,
                            float* __restrict__ C, int K, int Ncols, int colTiles) {
    const int lane = threadIdx.x & 31;
    const int wave = threadIdx.x >> 5;
    const int colTile = blockIdx.y * (blockDim.x >> 5) + wave;
    if (colTile >= colTiles) return;              // wave-uniform: EXEC stays all-ones
    const int row0 = blockIdx.x << 4;
    const int mrow = row0 + (lane & 15);
    const int ncol = (colTile << 4) + (lane & 15);
    const int hiK  = (lane >> 4) << 3;            // lanes 0-15: K 0..7/16..23; 16-31: 8..15/24..31

    const _Float16* __restrict__ arow = A  + (size_t)mrow * K;
    const _Float16* __restrict__ brow = BT + (size_t)ncol * K;

    v8f acc = {};
    for (int k0 = 0; k0 < K; k0 += 32) {
        half8 a_lo = *(const half8*)(arow + k0 + hiK);
        half8 a_hi = *(const half8*)(arow + k0 + 16 + hiK);
        half8 b_lo = *(const half8*)(brow + k0 + hiK);
        half8 b_hi = *(const half8*)(brow + k0 + 16 + hiK);
        v16h a = __builtin_shufflevector(a_lo, a_hi, 0,1,2,3,4,5,6,7,8,9,10,11,12,13,14,15);
        v16h b = __builtin_shufflevector(b_lo, b_hi, 0,1,2,3,4,5,6,7,8,9,10,11,12,13,14,15);
        acc = __builtin_amdgcn_wmma_f32_16x16x32_f16(
                  false, a, false, b, (short)0, acc, false, false);
    }

    // D layout: acc[j] -> row (row0 + 8*(lane>>4) + j), col = ncol
    const int rbase = row0 + ((lane >> 4) << 3);
    float* cptr = C + (size_t)rbase * Ncols + (colTile << 4) + (lane & 15);
#pragma unroll
    for (int j = 0; j < 8; ++j) cptr[(size_t)j * Ncols] = acc[j];
}

// ---------------------------------------------------------------------------
// el[n,h] = sum_d feat[n,h*D+d]*al[h,d] ; er likewise
// ---------------------------------------------------------------------------
__global__ void k_attn_proj(const float* __restrict__ feat, const float* __restrict__ al,
                            const float* __restrict__ ar, float* __restrict__ el,
                            float* __restrict__ er, int n, int H, int D) {
    int i = blockIdx.x * blockDim.x + threadIdx.x;
    if (i >= n * H) return;
    int node = i / H, h = i - node * H;
    const float* f = feat + (size_t)node * H * D + (size_t)h * D;
    const float* a_l = al + h * D;
    const float* a_r = ar + h * D;
    float sl = 0.f, sr = 0.f;
    for (int d = 0; d < D; ++d) { float v = f[d]; sl += v * a_l[d]; sr += v * a_r[d]; }
    el[i] = sl; er[i] = sr;
}

// agg[n,c] = rscale*res[n,c] + b[c]   (folds residual + bias into aggregation init)
__global__ void k_init_agg(const float* __restrict__ res, const float* __restrict__ b,
                           float* __restrict__ agg, long long total, int C, float rscale) {
    long long i = (long long)blockIdx.x * blockDim.x + threadIdx.x;
    if (i >= total) return;
    int c = (int)(i % C);
    agg[i] = rscale * res[i] + b[c];
}

// pass 1: ee = exp(leaky(el[src]+er[dst])); denom[dst,h] += ee   (softmax shift-invariant)
__global__ void k_edge1(const int* __restrict__ src, const int* __restrict__ dst,
                        const float* __restrict__ el, const float* __restrict__ er,
                        float* __restrict__ ee, float* __restrict__ denom, int E, int H) {
    long long i = (long long)blockIdx.x * blockDim.x + threadIdx.x;
    if (i >= (long long)E * H) return;
    int e = (int)(i / H), h = (int)(i - (long long)e * H);
    int s = src[e], d = dst[e];
    float v = el[(size_t)s * H + h] + er[(size_t)d * H + h];
    v = (v > 0.f) ? v : v * NEG_SLOPE;
    float ex = __expf(v);
    ee[i] = ex;
    atomicAdd(&denom[(size_t)d * H + h], ex);
}

// pass 2: agg[dst,c] += (ee/denom[dst,h]) * feat[src,c]
__global__ void k_edge2(const int* __restrict__ src, const int* __restrict__ dst,
                        const float* __restrict__ ee, const float* __restrict__ denom,
                        const float* __restrict__ feat, float* __restrict__ agg,
                        int E, int C, int D, int H) {
    long long i = (long long)blockIdx.x * blockDim.x + threadIdx.x;
    long long total = (long long)E * C;
    if (i >= total) return;
    int e = (int)(i / C);
    int c = (int)(i - (long long)e * C);
    int h = c / D;
    int s = src[e], d = dst[e];
    float dn = denom[(size_t)d * H + h];
    float alpha = ee[(size_t)e * H + h] / fmaxf(dn, 1e-16f);
    atomicAdd(&agg[(size_t)d * C + c], alpha * feat[(size_t)s * C + c]);
}

// BN stats: strided partial sums per channel, one atomic per thread per stat
__global__ void k_bn_stats(const float* __restrict__ x, float* __restrict__ sum,
                           float* __restrict__ sumsq, int n, int C) {
    int tid = blockIdx.x * blockDim.x + threadIdx.x;
    int c = tid % C;
    int r0 = tid / C;
    int rstride = (gridDim.x * blockDim.x) / C;
    float s = 0.f, s2 = 0.f;
    for (int r = r0; r < n; r += rstride) {
        float v = x[(size_t)r * C + c];
        s += v; s2 += v * v;
    }
    atomicAdd(&sum[c], s);
    atomicAdd(&sumsq[c], s2);
}

__global__ void k_bn_final(const float* __restrict__ sum, const float* __restrict__ sumsq,
                           const float* __restrict__ gamma, const float* __restrict__ beta,
                           float* __restrict__ scale, float* __restrict__ shift, int n, int C) {
    int c = blockIdx.x * blockDim.x + threadIdx.x;
    if (c >= C) return;
    float inv_n = 1.f / (float)n;
    float mu  = sum[c] * inv_n;
    float var = sumsq[c] * inv_n - mu * mu;
    float sc  = gamma[c] * rsqrtf(var + BN_EPS);
    scale[c] = sc;
    shift[c] = beta[c] - mu * sc;
}

// fused BN + ReLU, writes f32 result and f16 copy (feeds next layer's WMMA)
__global__ void k_bn_apply_relu(const float* __restrict__ x, const float* __restrict__ scale,
                                const float* __restrict__ shift, float* __restrict__ y,
                                _Float16* __restrict__ yH, long long total, int C) {
    long long i = (long long)blockIdx.x * blockDim.x + threadIdx.x;
    if (i >= total) return;
    int c = (int)(i % C);
    float v = fmaxf(x[i] * scale[c] + shift[c], 0.f);
    y[i] = v;
    yH[i] = (_Float16)v;
}

// ---------------------------------------------------------------------------
extern "C" void kernel_launch(void* const* d_in, const int* in_sizes, int n_in,
                              void* d_out, int out_size, void* d_ws, size_t ws_size,
                              hipStream_t stream) {
    const float* feat  = (const float*)d_in[0];
    const int*   srcI  = (const int*)d_in[1];
    const int*   dstI  = (const int*)d_in[2];
    const float* W0    = (const float*)d_in[3];
    const float* al0   = (const float*)d_in[4];
    const float* ar0   = (const float*)d_in[5];
    const float* b0    = (const float*)d_in[6];
    const float* resW0 = (const float*)d_in[7];
    const float* g0    = (const float*)d_in[8];
    const float* be0   = (const float*)d_in[9];
    const float* W1    = (const float*)d_in[10];
    const float* al1   = (const float*)d_in[11];
    const float* ar1   = (const float*)d_in[12];
    const float* b1    = (const float*)d_in[13];
    const float* g1    = (const float*)d_in[14];
    const float* be1   = (const float*)d_in[15];
    const float* W2    = (const float*)d_in[16];
    const float* al2   = (const float*)d_in[17];
    const float* ar2   = (const float*)d_in[18];
    const float* b2    = (const float*)d_in[19];
    const float* resW2 = (const float*)d_in[20];

    const int IN   = 128, H = 4, HID = 16, OUT = 40;
    const int Nn   = in_sizes[0] / IN;   // 100000 (multiple of 16)
    const int Ee   = in_sizes[1];        // 1600000
    const int C0   = H * HID;            // 64
    const int C2   = H * OUT;            // 160

    // ---- workspace layout (256B aligned slabs) ----
    char*  base = (char*)d_ws;
    size_t off  = 0;
    auto alloc = [&](size_t bytes) -> char* {
        char* p = base + off;
        off += (bytes + 255) & ~(size_t)255;
        return p;
    };
    _Float16* featH   = (_Float16*)alloc((size_t)Nn * IN * 2);
    _Float16* hH      = (_Float16*)alloc((size_t)Nn * C0 * 2);   // f16 activations (reused L1/L2)
    _Float16* W0T     = (_Float16*)alloc((size_t)IN * C0 * 2);
    _Float16* rW0T    = (_Float16*)alloc((size_t)IN * C0 * 2);
    _Float16* W1T     = (_Float16*)alloc((size_t)C0 * C0 * 2);
    _Float16* W2T     = (_Float16*)alloc((size_t)C0 * C2 * 2);
    _Float16* rW2T    = (_Float16*)alloc((size_t)C0 * C2 * 2);
    float* hfeat      = (float*)alloc((size_t)Nn * C2 * 4);      // projected features (max 160)
    float* resb       = (float*)alloc((size_t)Nn * C2 * 4);      // residual projections
    float* agg        = (float*)alloc((size_t)Nn * C0 * 4);
    float* h0         = (float*)alloc((size_t)Nn * C0 * 4);
    float* h1         = (float*)alloc((size_t)Nn * C0 * 4);
    float* el         = (float*)alloc((size_t)Nn * H * 4);
    float* er         = (float*)alloc((size_t)Nn * H * 4);
    float* denom      = (float*)alloc((size_t)Nn * H * 4);
    float* ee         = (float*)alloc((size_t)Ee * H * 4);
    float* bnsum      = (float*)alloc(C0 * 4);
    float* bnsumsq    = (float*)alloc(C0 * 4);
    float* bnscale    = (float*)alloc(C0 * 4);
    float* bnshift    = (float*)alloc(C0 * 4);

    float* outF = (float*)d_out;

    auto cdiv = [](long long a, long long b) { return (int)((a + b - 1) / b); };

    // ---- precision conversion (weights + input features) ----
    { long long t = (long long)Nn * IN;
      k_cvt_f16<<<cdiv(t,256),256,0,stream>>>(feat, featH, t); }
    k_cvt_wT<<<cdiv(IN*C0,256),256,0,stream>>>(W0,    W0T,  IN, C0);
    k_cvt_wT<<<cdiv(IN*C0,256),256,0,stream>>>(resW0, rW0T, IN, C0);
    k_cvt_wT<<<cdiv(C0*C0,256),256,0,stream>>>(W1,    W1T,  C0, C0);
    k_cvt_wT<<<cdiv(C0*C2,256),256,0,stream>>>(W2,    W2T,  C0, C2);
    k_cvt_wT<<<cdiv(C0*C2,256),256,0,stream>>>(resW2, rW2T, C0, C2);

    const int rowTiles = Nn / 16;   // 6250

    auto gemm = [&](const _Float16* A, const _Float16* BT, float* C, int K, int Ncols) {
        int colTiles = Ncols / 16;
        dim3 grid(rowTiles, (colTiles + 3) / 4);
        k_gemm_wmma<<<grid, 128, 0, stream>>>(A, BT, C, K, Ncols, colTiles);
    };

    // One GAT conv layer (after feat-projection GEMM done here)
    auto gat_edges = [&](const _Float16* Ah, int K, int D, int C,
                         const _Float16* WT, const float* al, const float* ar,
                         const float* bbias, const float* resbuf, float rscale,
                         float* aggbuf) {
        gemm(Ah, WT, hfeat, K, C);
        k_attn_proj<<<cdiv((long long)Nn*H,256),256,0,stream>>>(hfeat, al, ar, el, er, Nn, H, D);
        { long long t = (long long)Nn * C;
          k_init_agg<<<cdiv(t,256),256,0,stream>>>(resbuf, bbias, aggbuf, t, C, rscale); }
        hipMemsetAsync(denom, 0, (size_t)Nn * H * 4, stream);
        k_edge1<<<cdiv((long long)Ee*H,256),256,0,stream>>>(srcI, dstI, el, er, ee, denom, Ee, H);
        { long long t = (long long)Ee * C;
          k_edge2<<<cdiv(t,256),256,0,stream>>>(srcI, dstI, ee, denom, hfeat, aggbuf, Ee, C, D, H); }
    };

    auto batchnorm_relu = [&](const float* x, const float* gamma, const float* beta,
                              float* y, _Float16* yH) {
        hipMemsetAsync(bnsum,   0, C0 * 4, stream);
        hipMemsetAsync(bnsumsq, 0, C0 * 4, stream);
        k_bn_stats<<<128, 256, 0, stream>>>(x, bnsum, bnsumsq, Nn, C0);  // 32768 threads, /64 = 512 stripes
        k_bn_final<<<1, C0, 0, stream>>>(bnsum, bnsumsq, gamma, beta, bnscale, bnshift, Nn, C0);
        long long t = (long long)Nn * C0;
        k_bn_apply_relu<<<cdiv(t,256),256,0,stream>>>(x, bnscale, bnshift, y, yH, t, C0);
    };

    // ---- layer 0: res_fc = Linear(feat @ resW0); BN + ReLU ----
    gemm(featH, rW0T, resb, IN, C0);
    gat_edges(featH, IN, HID, C0, W0T, al0, ar0, b0, resb, 1.0f, agg);
    batchnorm_relu(agg, g0, be0, h0, hH);

    // ---- layer 1: identity res + outer residual fold to rscale=2; BN + ReLU ----
    gat_edges(hH, C0, HID, C0, W1T, al1, ar1, b1, h0, 2.0f, agg);
    batchnorm_relu(agg, g1, be1, h1, hH);   // hH now holds h1 in f16

    // ---- layer 2 (output): res_fc = Linear(h1 @ resW2); aggregate straight into d_out ----
    gemm(hH, rW2T, resb, C0, C2);
    gat_edges(hH, C0, OUT, C2, W2T, al2, ar2, b2, resb, 1.0f, outF);

    (void)n_in; (void)out_size; (void)ws_size; (void)h1;
}
